// DAEGC_73821897883963
// MI455X (gfx1250) — matrix operand order repaired and checked by reading, not compiled, not run on hardware
//
#include <hip/hip_runtime.h>
#include <hip/hip_bf16.h>

// ---------------------------------------------------------------------------
// GAT autoencoder (DAEGC-style) for MI455X / gfx1250, wave32 + WMMA f16.
// N=4096, F_IN=1024, HID=256, OUT=16.
// All GEMM operands are kept f16 row-major in memory (B pre-transposed), so
// WMMA fragments load DIRECTLY from global (no LDS round trip): per the ISA,
// row-major 16-bit fragments are plain GLOBAL_LOAD_B128 with per-lane
// addresses. Register blocking: 1 A-frag x 4 B-frags -> 4 v_wmma per K-step.
// Epilogues (ELU/sigmoid, f32/f16/transposed-f16 stores) fused into GEMMs.
// Bandwidth-bound: ~220MB HBM traffic -> ~10us at 23.3 TB/s.
// ---------------------------------------------------------------------------

typedef __attribute__((ext_vector_type(16))) _Float16 v16h;
typedef __attribute__((ext_vector_type(8)))  float    v8f;
typedef unsigned long long ull;

union V16H { v16h v; ull q[4]; };

#define GAT_N 4096

// ---- WMMA fragment loads, straight from global f16 row-major --------------
// A [M,K]: lane l<16 -> M=l, a[0..15]=K0..15 ;
//          lane l>=16 -> M=l-16, a[0..15]=K16..31.
__device__ inline v16h frag_a_g(const _Float16* __restrict__ A, int lda,
                                int m0, int k0, int lane) {
  const int row = lane & 15, hi = lane >> 4;
  const _Float16* p = A + (size_t)(m0 + row) * lda + k0 + hi * 16;
  V16H r;
  const ull* q0 = (const ull*)p;          // 16B-aligned chunk (K blk 0)
  const ull* q1 = (const ull*)(p + 8);    // 16B-aligned chunk (K blk 1)
  r.q[0] = q0[0]; r.q[1] = q0[1];
  r.q[2] = q1[0]; r.q[3] = q1[1];
  return r.v;
}

// B given pre-transposed as Bt [N,K] row-major: lane l -> N=l&15,
// b[i] = B[K=(l>>4)*16+i][N]  ->  32 contiguous bytes of Bt row.
__device__ inline v16h frag_b_g(const _Float16* __restrict__ Bt, int ldb,
                                int n0, int k0, int lane) {
  const int n = lane & 15, kh = lane >> 4;
  const ull* q = (const ull*)(Bt + (size_t)(n0 + n) * ldb + k0 + kh * 16);
  V16H r;
  r.q[0] = q[0]; r.q[1] = q[1]; r.q[2] = q[2]; r.q[3] = q[3];
  return r.v;
}

// ---- register-blocked WMMA GEMM: D = epi(A @ Bt^T) ------------------------
// A [M,K] f16 (lda), Bt [N,K] f16 (ldb). Each wave: 16 x (16*NBLK) output.
// epi: 0=none, 1=ELU, 2=sigmoid. Optional stores: C (f32), C16 (f16),
// CT16 (f16 transposed [N,M]) -- producers emit the operand layout the next
// GEMM wants, so every downstream fragment load is contiguous.
template <int NBLK>
__global__ __launch_bounds__(128)
void wmma_gemm(const _Float16* __restrict__ A, const _Float16* __restrict__ Bt,
               float* __restrict__ C, _Float16* __restrict__ C16,
               _Float16* __restrict__ CT16,
               int M, int N, int K, int lda, int ldb, int ldc, int epi) {
  const int lane = threadIdx.x & 31;
  const int wtile = blockIdx.x * 4 + (threadIdx.x >> 5);
  const int nBlocks = N / (16 * NBLK);
  if (wtile >= (M >> 4) * nBlocks) return;            // wave-uniform
  const int tM  = wtile / nBlocks;
  const int tN0 = (wtile % nBlocks) * NBLK;

  v8f acc[NBLK] = {};

#pragma unroll 2
  for (int k0 = 0; k0 < K; k0 += 32) {
    const v16h a = frag_a_g(A, lda, tM * 16, k0, lane);   // reused NBLK times
#pragma unroll
    for (int i = 0; i < NBLK; ++i) {
      const v16h b = frag_b_g(Bt, ldb, (tN0 + i) * 16, k0, lane);
      acc[i] = __builtin_amdgcn_wmma_f32_16x16x32_f16(
          /*neg_a=*/false, a, /*neg_b=*/false, b,
          /*c_mod=*/(short)0, acc[i], /*reuse_a=*/false, /*reuse_b=*/false);
    }
    if (k0 + 32 < K)                                    // global_prefetch_b8
      __builtin_prefetch(A + (size_t)(tM * 16 + (lane & 15)) * lda + k0 + 32,
                         0, 1);
  }

  // C/D layout: lane col = lane&15, VGPR v holds row v + 8*(lane>=16)
  const int col = lane & 15, hi = lane >> 4;
#pragma unroll
  for (int i = 0; i < NBLK; ++i) {
    const int n = (tN0 + i) * 16 + col;
#pragma unroll
    for (int v = 0; v < 8; ++v) {
      const int mrow = tM * 16 + v + 8 * hi;
      float x = acc[i][v];
      if (epi == 1)      x = (x > 0.f) ? x : (__expf(x) - 1.f);   // ELU
      else if (epi == 2) x = 1.f / (1.f + __expf(-x));            // sigmoid
      if (C)    C  [(size_t)mrow * ldc + n] = x;
      if (C16)  C16[(size_t)mrow * ldc + n] = (_Float16)x;
      if (CT16) CT16[(size_t)n * M + mrow]  = (_Float16)x;  // contiguous in v
    }
  }
}

// ---- elementwise f32 -> f16 convert ---------------------------------------
__global__ __launch_bounds__(256)
void cvt16_kernel(const float* __restrict__ in, _Float16* __restrict__ out,
                  size_t n) {
  size_t i = (size_t)blockIdx.x * 256 + threadIdx.x;
  if (i < n) out[i] = (_Float16)in[i];
}

// ---- transpose + convert: in [R,C] f32 -> out [C,R] f16 -------------------
__global__ __launch_bounds__(256)
void transpose_cvt_kernel(const float* __restrict__ in,
                          _Float16* __restrict__ out, int R, int Ccols) {
  int idx = blockIdx.x * 256 + threadIdx.x;
  if (idx < R * Ccols) {
    int r = idx / Ccols, c = idx % Ccols;
    out[(size_t)c * R + r] = (_Float16)in[idx];
  }
}

// ---- per-row dual dot products: s[i]=h[i]·a_s, n[i]=h[i]·a_n --------------
__global__ __launch_bounds__(256)
void dots_kernel(const float* __restrict__ H, int d,
                 const float* __restrict__ a_s, const float* __restrict__ a_n,
                 float* __restrict__ s, float* __restrict__ n, int rows) {
  const int wave = threadIdx.x >> 5, lane = threadIdx.x & 31;
  const int row = blockIdx.x * 8 + wave;
  if (row >= rows) return;
  const float* h = H + (size_t)row * d;
  float ss = 0.f, nn = 0.f;
  for (int c = lane; c < d; c += 32) {
    float hv = h[c];
    ss = fmaf(hv, a_s[c], ss);
    nn = fmaf(hv, a_n[c], nn);
  }
#pragma unroll
  for (int off = 16; off; off >>= 1) {
    ss += __shfl_xor(ss, off, 32);
    nn += __shfl_xor(nn, off, 32);
  }
  if (lane == 0) { s[row] = ss; n[row] = nn; }
}

// ---- masked attention row softmax, single pass, f16 output ----------------
// e[i,j] = leaky_relu(m[i,j]*(s[i]+n[j]), 0.2) masked to -9e15 off-edges.
// 4096 cols / 256 threads = 16 e-values per thread held in VGPRs: one read
// sweep of m/adj, one f16 write of att. No intermediate memory traffic.
__global__ __launch_bounds__(256)
void attn_softmax_kernel(const float* __restrict__ m,
                         const float* __restrict__ adj,
                         const float* __restrict__ s,
                         const float* __restrict__ n,
                         _Float16* __restrict__ att, int N) {
  const int row = blockIdx.x;
  const int tid = threadIdx.x, lane = tid & 31, wave = tid >> 5;
  __shared__ float red[8];
  const float si = s[row];
  const float* mr = m   + (size_t)row * N;
  const float* ar = adj + (size_t)row * N;

  float ev[16];
  float mx = -3.0e38f;
#pragma unroll
  for (int t = 0; t < 16; ++t) {
    const int j = tid + t * 256;
    float e = (si + n[j]) * mr[j];
    e = (e > 0.f) ? e : 0.2f * e;
    e = (ar[j] > 0.f) ? e : -9.0e15f;
    ev[t] = e;
    mx = fmaxf(mx, e);
  }
#pragma unroll
  for (int off = 16; off; off >>= 1) mx = fmaxf(mx, __shfl_xor(mx, off, 32));
  if (lane == 0) red[wave] = mx;
  __syncthreads();
  mx = red[0];
#pragma unroll
  for (int w = 1; w < 8; ++w) mx = fmaxf(mx, red[w]);
  __syncthreads();

  float sum = 0.f;
#pragma unroll
  for (int t = 0; t < 16; ++t) {
    const float p = __expf(ev[t] - mx);
    ev[t] = p;
    sum += p;
  }
#pragma unroll
  for (int off = 16; off; off >>= 1) sum += __shfl_xor(sum, off, 32);
  if (lane == 0) red[wave] = sum;
  __syncthreads();
  float tot = 0.f;
#pragma unroll
  for (int w = 0; w < 8; ++w) tot += red[w];
  const float inv = 1.f / tot;
#pragma unroll
  for (int t = 0; t < 16; ++t)
    att[(size_t)row * N + tid + t * 256] = (_Float16)(ev[t] * inv);
}

// ---- row L2-normalize (OUT=16): f32 z to d_out + zero-padded f16 [N,32] ---
__global__ __launch_bounds__(256)
void norm_kernel(const float* __restrict__ h, float* __restrict__ zout,
                 _Float16* __restrict__ z16, int rows) {
  const int wave = threadIdx.x >> 5, lane = threadIdx.x & 31;
  const int row = blockIdx.x * 8 + wave;
  if (row >= rows) return;
  float v = (lane < 16) ? h[(size_t)row * 16 + lane] : 0.f;
  float sq = v * v;
#pragma unroll
  for (int off = 16; off; off >>= 1) sq += __shfl_xor(sq, off, 32);
  const float inv = 1.f / fmaxf(sqrtf(sq), 1e-12f);
  const float z = v * inv;
  if (lane < 16) zout[(size_t)row * 16 + lane] = z;
  z16[(size_t)row * 32 + lane] = (_Float16)((lane < 16) ? z : 0.f);
}

// ---------------------------------------------------------------------------
extern "C" void kernel_launch(void* const* d_in, const int* in_sizes, int n_in,
                              void* d_out, int out_size, void* d_ws, size_t ws_size,
                              hipStream_t stream) {
  (void)in_sizes; (void)n_in; (void)out_size; (void)ws_size;
  const int N = GAT_N;
  const float* x   = (const float*)d_in[0];   // [4096,1024]
  const float* adj = (const float*)d_in[1];   // [4096,4096]
  const float* m   = (const float*)d_in[2];   // [4096,4096]
  const float* W1  = (const float*)d_in[3];   // [1024,256]
  const float* as1 = (const float*)d_in[4];   // [256]
  const float* an1 = (const float*)d_in[5];   // [256]
  const float* W2  = (const float*)d_in[6];   // [256,16]
  const float* as2 = (const float*)d_in[7];   // [16]
  const float* an2 = (const float*)d_in[8];   // [16]

  // ---- workspace carve (256B aligned), ~50 MB total ----
  char* wp = (char*)d_ws;
  auto carve = [&](size_t bytes) {
    void* r = wp;
    wp += (bytes + 255) & ~(size_t)255;
    return r;
  };
  _Float16* att16 = (_Float16*)carve((size_t)N * N * 2);        // 32 MB
  _Float16* x16   = (_Float16*)carve((size_t)N * 1024 * 2);     // 8 MB
  _Float16* W1T   = (_Float16*)carve((size_t)256 * 1024 * 2);   // [256,1024]
  _Float16* W2T   = (_Float16*)carve((size_t)16 * 256 * 2);     // [16,256]
  float*    h1    = (float*)   carve((size_t)N * 256 * 4);      // 4 MB
  _Float16* h1T   = (_Float16*)carve((size_t)256 * N * 2);      // [256,4096]
  _Float16* h1p16 = (_Float16*)carve((size_t)N * 256 * 2);      // [4096,256]
  float*    h2    = (float*)   carve((size_t)N * 16 * 4);
  _Float16* h2T   = (_Float16*)carve((size_t)16 * N * 2);       // [16,4096]
  float*    h2p   = (float*)   carve((size_t)N * 16 * 4);
  _Float16* z16   = (_Float16*)carve((size_t)N * 32 * 2);       // K-padded
  float*    s1    = (float*)   carve((size_t)N * 4);
  float*    n1    = (float*)   carve((size_t)N * 4);
  float*    s2    = (float*)   carve((size_t)N * 4);
  float*    n2    = (float*)   carve((size_t)N * 4);

  float* adj_pred = (float*)d_out;                   // [N,N]
  float* zout     = (float*)d_out + (size_t)N * N;   // [N,16]

  _Float16* const F16NULL = nullptr;
  float*    const F32NULL = nullptr;

  // ---- operand preparation ----
  cvt16_kernel<<<(N * 1024) / 256, 256, 0, stream>>>(x, x16, (size_t)N * 1024);
  transpose_cvt_kernel<<<(1024 * 256) / 256, 256, 0, stream>>>(W1, W1T, 1024, 256);
  transpose_cvt_kernel<<<(256 * 16 + 255) / 256, 256, 0, stream>>>(W2, W2T, 256, 16);

  // ---- layer 1 ----
  // h1 = x @ W1 : f32 for dot products + transposed f16 for next GEMM's B
  wmma_gemm<4><<<(256 * 4) / 4, 128, 0, stream>>>(
      x16, W1T, h1, F16NULL, h1T, N, 256, 1024, 1024, 1024, 256, 0);
  dots_kernel<<<N / 8, 256, 0, stream>>>(h1, 256, as1, an1, s1, n1, N);
  attn_softmax_kernel<<<N, 256, 0, stream>>>(m, adj, s1, n1, att16, N);
  // h1' = elu(att @ h1) : f16 row-major (A operand of next GEMM)
  wmma_gemm<4><<<(256 * 4) / 4, 128, 0, stream>>>(
      att16, h1T, F32NULL, h1p16, F16NULL, N, 256, N, N, N, 256, 1);

  // ---- layer 2 ----
  // h2 = h1' @ W2 : f32 for dots + transposed f16 for aggregation B
  wmma_gemm<1><<<256 / 4, 128, 0, stream>>>(
      h1p16, W2T, h2, F16NULL, h2T, N, 16, 256, 256, 256, 16, 0);
  dots_kernel<<<N / 8, 256, 0, stream>>>(h2, 16, as2, an2, s2, n2, N);
  attn_softmax_kernel<<<N, 256, 0, stream>>>(m, adj, s2, n2, att16, N);
  // h2' = elu(att @ h2) : f32 for the normalizer
  wmma_gemm<1><<<256 / 4, 128, 0, stream>>>(
      att16, h2T, h2p, F16NULL, F16NULL, N, 16, N, N, N, 16, 1);

  // ---- normalize + decoder ----
  norm_kernel<<<N / 8, 256, 0, stream>>>(h2p, zout, z16, N);
  // adj_pred = sigmoid(z @ z^T) ; z16 is both A [N,32] and Bt [N,32]
  wmma_gemm<4><<<(256 * 64) / 4, 128, 0, stream>>>(
      z16, z16, adj_pred, F16NULL, F16NULL, N, N, 32, 32, 32, N, 2);
}
